// AutopoieticEngine_37752762532596
// MI455X (gfx1250) — matrix operand order repaired and checked by reading, not compiled, not run on hardware
//
#include <hip/hip_runtime.h>
#include <hip/hip_bf16.h>
#include <cstdint>

// ---------------------------------------------------------------------------
// MI455X (gfx1250) implementation.
// Matmuls in bf16 WMMA (v_wmma_f32_16x16x32_bf16), f32 accumulate; all
// elementwise GRU / softmax math in f32. Memory-bound target (~270MB HBM
// traffic @ 23.3 TB/s ~ 12us); bf16 WMMA keeps compute under that.
// Weight staging into LDS uses the CDNA5 Tensor Data Mover (tensor_load_to_lds
// + s_wait_tensorcnt).
// ---------------------------------------------------------------------------

typedef __bf16 bf16_t;
typedef __attribute__((ext_vector_type(16))) __bf16 v16bf;
typedef __attribute__((ext_vector_type(8)))  __bf16 v8bf;
typedef __attribute__((ext_vector_type(8)))  float  v8f;
typedef uint32_t u32x4 __attribute__((ext_vector_type(4)));
typedef uint32_t u32x8 __attribute__((ext_vector_type(8)));

#define HID 128
#define IND 64
#define CD  192   // IND + HID
#define OUTD 64
#define KIH 96    // 65 padded to 3*32

// ---------------- weight packing: W(Nout x Kin) f32 -> B-fragment bf16 ------
// fragment = 32(K) x 16(N) tile; per-lane 16 bf16 contiguous:
//   lane l: N = nt*16 + (l&15), elements e: K = kt*32 + 16*(l>>4) + e
// fragment order: frag = nt*KT + kt
__global__ void pack_w_k(const float* __restrict__ W, bf16_t* __restrict__ P,
                         int Kin, int Nout, int KT, int NT) {
  int total = KT * NT * 512;
  for (int p = blockIdx.x * blockDim.x + threadIdx.x; p < total;
       p += gridDim.x * blockDim.x) {
    int frag = p >> 9, r = p & 511;
    int ln = r >> 4, e = r & 15;
    int nt = frag / KT, kt = frag - nt * KT;
    int N = nt * 16 + (ln & 15);
    int K = kt * 32 + ((ln >> 4) << 4) + e;
    float v = (K < Kin && N < Nout) ? W[(long)N * Kin + K] : 0.0f;
    P[p] = (bf16_t)v;
  }
}

// ---------------- fragment loaders -----------------------------------------
// A fragment (16x32, M x K), contiguous k matching pack_w_k: lane l row M=l&15; elements:
//   K = k0 + 16*(l>>4) + e
__device__ __forceinline__ v16bf load_afrag(const bf16_t* base, int stride,
                                            int k0, int lane) {
  int m = lane & 15, g = lane >> 4;
  const bf16_t* p = base + m * stride + k0 + g * 16;
  v8bf lo = *(const v8bf*)p;
  v8bf hi = *(const v8bf*)(p + 8);
  v16bf a;
#pragma unroll
  for (int e = 0; e < 8; ++e) { a[e] = lo[e]; a[8 + e] = hi[e]; }
  return a;
}

__device__ __forceinline__ v16bf load_bfrag(const bf16_t* pack, int frag,
                                            int lane) {
  return *(const v16bf*)(pack + frag * 512 + lane * 16);
}

__device__ __forceinline__ v8f wmma_bf(v16bf a, v16bf b, v8f c) {
  return __builtin_amdgcn_wmma_f32_16x16x32_bf16(
      false, a, false, b, (short)0, c, false, false);
}

__device__ __forceinline__ void lds_fence() {
  asm volatile("s_wait_dscnt 0" ::: "memory");
}

// ---- Tensor Data Mover: contiguous global -> LDS DMA (1-row 2D tile) ------
// D# per CDNA5 ISA: group0 = {count/flags, lds_addr,
// global_addr lo, global_addr hi | type=2}; group1 = {data_size=8B,
// tensor_dim0 = tile_dim0 = bytes/8, tensor_dim1 = tile_dim1 = 1,
// stride0 = bytes/8}. Tracked with TENSORcnt.
__device__ __forceinline__ void tdm_load_1d(const void* gsrc, void* lds_dst,
                                            uint32_t bytes) {
  uint64_t ga = (uint64_t)(uintptr_t)gsrc;
  uint32_t la = (uint32_t)(uintptr_t)lds_dst;   // generic lo 32 = LDS offset
  uint32_t n8 = bytes >> 3;                     // 8-byte elements (<= 65535)
  u32x4 g0;
  g0[0] = 1u;                                            // count=1, user D#
  g0[1] = la;                                            // lds_addr
  g0[2] = (uint32_t)ga;                                  // global_addr[31:0]
  g0[3] = (uint32_t)((ga >> 32) & 0x01FFFFFFu) | (2u << 30); // [56:32]|type=2
  u32x8 g1;
  g1[0] = 3u << 16;                    // data_size = 8 bytes
  g1[1] = (n8 & 0xFFFFu) << 16;        // tensor_dim0[15:0]
  g1[2] = (n8 >> 16) | (1u << 16);     // tensor_dim0[31:16] | tensor_dim1 = 1
  g1[3] = (n8 & 0xFFFFu) << 16;        // tile_dim0 = n8
  g1[4] = 1u;                          // tile_dim1 = 1, tile_dim2 = 0
  g1[5] = n8;                          // tensor_dim0_stride[31:0]
  g1[6] = 0u;                          // stride0 hi | stride1 lo
  g1[7] = 0u;                          // stride1 hi
  asm volatile("tensor_load_to_lds %0, %1" :: "s"(g0), "s"(g1) : "memory");
}

// ---------------- zero accumulators ----------------------------------------
__global__ void zero_k(float* fsum, float* numer, float* denom, float* tsum,
                       int* tmaxI) {
  int t = threadIdx.x;
  if (t < 1024) fsum[t] = 0.f;
  if (t < 64) numer[t] = 0.f;
  if (t == 0) { *denom = 0.f; *tsum = 0.f; *tmaxI = 0; }
}

// ---------------- main per-row kernel --------------------------------------
// 256 threads = 8 wave32; each wave owns a 16-row tile (WG = 128 rows, all in
// one faction since fs is a multiple of 128).
__global__ void __launch_bounds__(256, 1)
main_k(const float* __restrict__ x, const float* __restrict__ hiddens,
       const float* __restrict__ ba1, const float* __restrict__ ba2,
       const float* __restrict__ bg1, const float* __restrict__ bg2,
       const float* __restrict__ b_ih, const float* __restrict__ b_hh,
       const int* __restrict__ aidx,
       const bf16_t* __restrict__ pWa1, const bf16_t* __restrict__ pWg1,
       const bf16_t* __restrict__ pWa2, const bf16_t* __restrict__ pWg2,
       const bf16_t* __restrict__ pWih, const bf16_t* __restrict__ pWhh,
       float* __restrict__ out_ws, float* __restrict__ t_ws,
       float* __restrict__ gFsum, float* __restrict__ gTsum,
       int* __restrict__ gTmaxI, float* __restrict__ doutH, int fs) {
  __shared__ __align__(32) bf16_t sWa1[48 * 512];   //  48 KB
  __shared__ __align__(32) bf16_t sWg1[48 * 512];   //  48 KB
  __shared__ __align__(32) bf16_t sWhh[96 * 512];   //  96 KB
  __shared__ __align__(32) bf16_t sC[8][16 * CD];   //  48 KB  c = [x | h] bf16
  __shared__ __align__(32) bf16_t sAct[8][16 * HID];//  32 KB  relu / mem_in
  __shared__ float sFsum[128];
  __shared__ float sTsum;
  __shared__ int   sTmax;

  const int lane = threadIdx.x & 31;
  const int wv = threadIdx.x >> 5;

  // --- stage packed weights into LDS via Tensor Data Mover (wave 0) ---
  if (threadIdx.x < 32) {
    tdm_load_1d(pWa1, sWa1, 48 * 1024);
    tdm_load_1d(pWg1, sWg1, 48 * 1024);
    tdm_load_1d(pWhh, sWhh, 96 * 1024);
    __builtin_amdgcn_s_wait_tensorcnt(0);
  }
  if (threadIdx.x < 128) sFsum[threadIdx.x] = 0.f;
  if (threadIdx.x == 0) { sTsum = 0.f; sTmax = 0; }
  __syncthreads();

  const int r0 = (blockIdx.x * 8 + wv) * 16;
  int myrow = (lane < 16) ? aidx[r0 + lane] : 0;
  __builtin_prefetch(&aidx[r0 + 128], 0, 1);  // global_prefetch_b8

  bf16_t* cw = &sC[wv][0];
  bf16_t* aw = &sAct[wv][0];

  // --- stage c = [x | h] as bf16 (64-bit packed LDS stores) ---
#pragma unroll
  for (int i = 0; i < 8; ++i) {               // x: 16 rows x 16 float4
    int idx = i * 32 + lane, row = idx >> 4, c4 = idx & 15;
    float4 v = ((const float4*)x)[c4];
    union { bf16_t h[4]; unsigned long long u; } q;
    q.h[0] = (bf16_t)v.x; q.h[1] = (bf16_t)v.y;
    q.h[2] = (bf16_t)v.z; q.h[3] = (bf16_t)v.w;
    *(unsigned long long*)(cw + row * CD + c4 * 4) = q.u;
  }
#pragma unroll
  for (int i = 0; i < 16; ++i) {              // h: 16 rows x 32 float4
    int idx = i * 32 + lane, row = idx >> 5, c4 = idx & 31;
    int gr = __shfl(myrow, row, 32);
    float4 v = ((const float4*)(hiddens + (long)gr * HID))[c4];
    union { bf16_t h[4]; unsigned long long u; } q;
    q.h[0] = (bf16_t)v.x; q.h[1] = (bf16_t)v.y;
    q.h[2] = (bf16_t)v.z; q.h[3] = (bf16_t)v.w;
    *(unsigned long long*)(cw + row * CD + IND + c4 * 4) = q.u;
  }
  lds_fence();

  const int cN = lane & 15;       // N within tile
  const int half = lane >> 4;     // C-layout row group: M = j + 8*half

  // ================= branch A: relu(c@Wa1.T+ba1) @ Wa2.T ==================
#pragma unroll
  for (int nt = 0; nt < 8; ++nt) {
    v8f acc = {0, 0, 0, 0, 0, 0, 0, 0};
#pragma unroll
    for (int kt = 0; kt < 6; ++kt)
      acc = wmma_bf(load_afrag(cw, CD, kt * 32, lane),
                    load_bfrag(sWa1, nt * 6 + kt, lane), acc);
    float b1 = ba1[nt * 16 + cN];
#pragma unroll
    for (int j = 0; j < 8; ++j) {
      float v = acc[j] + b1; v = v > 0.f ? v : 0.f;
      aw[(j + 8 * half) * HID + nt * 16 + cN] = (bf16_t)v;
    }
  }
  lds_fence();
  v8f oA[4];
#pragma unroll
  for (int nt = 0; nt < 4; ++nt) {
    v8f acc = {0, 0, 0, 0, 0, 0, 0, 0};
#pragma unroll
    for (int kt = 0; kt < 4; ++kt)
      acc = wmma_bf(load_afrag(aw, HID, kt * 32, lane),
                    load_bfrag(pWa2, nt * 4 + kt, lane), acc);
    oA[nt] = acc;
  }

  // ================= branch G =============================================
#pragma unroll
  for (int nt = 0; nt < 8; ++nt) {
    v8f acc = {0, 0, 0, 0, 0, 0, 0, 0};
#pragma unroll
    for (int kt = 0; kt < 6; ++kt)
      acc = wmma_bf(load_afrag(cw, CD, kt * 32, lane),
                    load_bfrag(sWg1, nt * 6 + kt, lane), acc);
    float b1 = bg1[nt * 16 + cN];
#pragma unroll
    for (int j = 0; j < 8; ++j) {
      float v = acc[j] + b1; v = v > 0.f ? v : 0.f;
      aw[(j + 8 * half) * HID + nt * 16 + cN] = (bf16_t)v;
    }
  }
  lds_fence();
  v8f outf[4];
  float rp[8];
#pragma unroll
  for (int j = 0; j < 8; ++j) rp[j] = 0.f;
#pragma unroll
  for (int nt = 0; nt < 4; ++nt) {
    v8f acc = {0, 0, 0, 0, 0, 0, 0, 0};
#pragma unroll
    for (int kt = 0; kt < 4; ++kt)
      acc = wmma_bf(load_afrag(aw, HID, kt * 32, lane),
                    load_bfrag(pWg2, nt * 4 + kt, lane), acc);
    float bd = ba2[nt * 16 + cN] - bg2[nt * 16 + cN];
#pragma unroll
    for (int j = 0; j < 8; ++j) {
      float o = oA[nt][j] - acc[j] + bd;
      outf[nt][j] = o;
      rp[j] += o * o;
    }
  }

  // tension per row: reduce over 16 lane-columns within each half-wave
#pragma unroll
  for (int m = 1; m < 16; m <<= 1)
#pragma unroll
    for (int j = 0; j < 8; ++j) rp[j] += __shfl_xor(rp[j], m, 32);
  float tj[8];
#pragma unroll
  for (int j = 0; j < 8; ++j) tj[j] = rp[j] * (1.0f / OUTD);

  // store out (f32) to ws + stage mem_in = [bf16(out), tension, 0-pad] in sAct
#pragma unroll
  for (int nt = 0; nt < 4; ++nt)
#pragma unroll
    for (int j = 0; j < 8; ++j) {
      int m = j + 8 * half;
      out_ws[(long)(r0 + m) * OUTD + nt * 16 + cN] = outf[nt][j];
      aw[m * HID + nt * 16 + cN] = (bf16_t)outf[nt][j];
    }
#pragma unroll
  for (int j = 0; j < 8; ++j) {
    int m = j + 8 * half;
    aw[m * HID + 64 + cN] = (cN == 0) ? (bf16_t)tj[j] : (bf16_t)0.f;
    aw[m * HID + 80 + cN] = (bf16_t)0.f;
  }
  if (cN == 0) {
    float ls = 0.f, lm = 0.f;
#pragma unroll
    for (int j = 0; j < 8; ++j) {
      int m = j + 8 * half;
      t_ws[r0 + m] = tj[j];
      ls += tj[j];
      lm = fmaxf(lm, tj[j]);
    }
    atomicAdd(&sTsum, ls);
    atomicMax(&sTmax, __float_as_int(lm));
  }
  lds_fence();

  // ================= GRU: gi = mem_in@W_ih.T, gh = h@W_hh.T ===============
#pragma unroll 1
  for (int t = 0; t < 8; ++t) {
    v8f gir = {0,0,0,0,0,0,0,0}, giz = {0,0,0,0,0,0,0,0}, gin = {0,0,0,0,0,0,0,0};
    v8f ghr = {0,0,0,0,0,0,0,0}, ghz = {0,0,0,0,0,0,0,0}, ghn = {0,0,0,0,0,0,0,0};
#pragma unroll
    for (int kt = 0; kt < 3; ++kt) {
      v16bf a = load_afrag(aw, HID, kt * 32, lane);
      gir = wmma_bf(a, load_bfrag(pWih, (t)      * 3 + kt, lane), gir);
      giz = wmma_bf(a, load_bfrag(pWih, (t + 8)  * 3 + kt, lane), giz);
      gin = wmma_bf(a, load_bfrag(pWih, (t + 16) * 3 + kt, lane), gin);
    }
#pragma unroll
    for (int kt = 0; kt < 4; ++kt) {
      v16bf a = load_afrag(cw, CD, IND + kt * 32, lane);
      ghr = wmma_bf(a, load_bfrag(sWhh, (t)      * 4 + kt, lane), ghr);
      ghz = wmma_bf(a, load_bfrag(sWhh, (t + 8)  * 4 + kt, lane), ghz);
      ghn = wmma_bf(a, load_bfrag(sWhh, (t + 16) * 4 + kt, lane), ghn);
    }
    int col = t * 16 + cN;
    float bir = b_ih[col], biz = b_ih[col + 128], bin = b_ih[col + 256];
    float bhr = b_hh[col], bhz = b_hh[col + 128], bhn = b_hh[col + 256];
    float fcol = 0.f;
#pragma unroll
    for (int j = 0; j < 8; ++j) {
      int m = j + 8 * half;
      int gr = __shfl(myrow, m, 32);
      float h = hiddens[(long)gr * HID + col];        // original f32 h
      float r = 1.f / (1.f + __expf(-(gir[j] + bir + ghr[j] + bhr)));
      float z = 1.f / (1.f + __expf(-(giz[j] + biz + ghz[j] + bhz)));
      float nn = tanhf(gin[j] + bin + r * (ghn[j] + bhn));
      float nh = (1.f - z) * nn + z * h;
      doutH[(long)gr * HID + col] = nh;               // pre-sync new_h
      fcol += nh;
    }
    fcol += __shfl_xor(fcol, 16, 32);                 // fold both halves
    if (lane < 16) atomicAdd(&sFsum[col], fcol);
  }

  __syncthreads();
  if (threadIdx.x < 128) {
    int f = (blockIdx.x * 128) / fs;
    atomicAdd(&gFsum[f * 128 + threadIdx.x], sFsum[threadIdx.x]);
  }
  if (threadIdx.x == 0) {
    atomicAdd(gTsum, sTsum);
    atomicMax(gTmaxI, sTmax);
  }
}

// ---------------- faction means + global debate vector ---------------------
__global__ void faction_k(const float* __restrict__ fsum, float* __restrict__ fm,
                          float* __restrict__ go, int fs) {
  int tid = threadIdx.x;
  if (tid < 128) {
    float s = 0.f;
    for (int f = 0; f < 8; ++f) {
      float m = fsum[f * 128 + tid] / (float)fs;
      fm[f * 128 + tid] = m;
      s += m;
    }
    go[tid] = s * 0.125f;
  }
}

// ---------------- sync/debate, in-place on scattered rows of d_out ---------
__global__ void sync_k(float* __restrict__ doutH, const int* __restrict__ aidx,
                       const float* __restrict__ fm, const float* __restrict__ go,
                       const int* __restrict__ step, int n, int fs, int dc) {
  long idx = (long)blockIdx.x * blockDim.x + threadIdx.x;
  if (idx >= (long)n * 32) return;
  int i = (int)(idx >> 5), c4 = (int)(idx & 31);
  int f = i / fs, pos = i - f * fs;
  int gr = aidx[i];
  float4* p = (float4*)(doutH + (long)gr * HID + c4 * 4);
  float4 v = *p;
  float4 fv = *(const float4*)(fm + f * 128 + c4 * 4);
  v.x = 0.85f * v.x + 0.15f * fv.x;
  v.y = 0.85f * v.y + 0.15f * fv.y;
  v.z = 0.85f * v.z + 0.15f * fv.z;
  v.w = 0.85f * v.w + 0.15f * fv.w;
  if (*step > 5 && pos < dc) {
    float4 gv = *(const float4*)(go + c4 * 4);
    v.x = 0.85f * v.x + 0.15f * gv.x;
    v.y = 0.85f * v.y + 0.15f * gv.y;
    v.z = 0.85f * v.z + 0.15f * gv.z;
    v.w = 0.85f * v.w + 0.15f * gv.w;
  }
  *p = v;
}

// ---------------- softmax-weighted combine ---------------------------------
__global__ void combine_k(const float* __restrict__ out_ws,
                          const float* __restrict__ t_ws,
                          const int* __restrict__ tmaxI,
                          float* __restrict__ numer, float* __restrict__ denom,
                          int n) {
  __shared__ float red[256];
  __shared__ float dred[4];
  int c = threadIdx.x & 63, g = threadIdx.x >> 6;
  float tmax = __int_as_float(*tmaxI);
  float acc = 0.f, dacc = 0.f;
  for (int i = blockIdx.x * 4 + g; i < n; i += gridDim.x * 4) {
    float e = __expf(t_ws[i] - tmax);
    acc += e * out_ws[(long)i * OUTD + c];
    if (c == 0) dacc += e;
  }
  red[threadIdx.x] = acc;
  if (c == 0) dred[g] = dacc;
  __syncthreads();
  if (threadIdx.x < 64) {
    float s = red[c] + red[64 + c] + red[128 + c] + red[192 + c];
    atomicAdd(&numer[c], s);
    if (c == 0) atomicAdd(denom, dred[0] + dred[1] + dred[2] + dred[3]);
  }
}

__global__ void final_k(float* __restrict__ dout, const float* __restrict__ numer,
                        const float* __restrict__ denom,
                        const float* __restrict__ tsum, int n) {
  int t = threadIdx.x;
  if (t < 64) dout[t] = numer[t] / *denom;
  if (t == 64) dout[64] = *tsum / (float)n;
}

// ---------------------------------------------------------------------------
extern "C" void kernel_launch(void* const* d_in, const int* in_sizes, int n_in,
                              void* d_out, int out_size, void* d_ws,
                              size_t ws_size, hipStream_t stream) {
  const float* x    = (const float*)d_in[0];
  const float* hid  = (const float*)d_in[1];
  const float* Wa1  = (const float*)d_in[2];
  const float* ba1  = (const float*)d_in[3];
  const float* Wa2  = (const float*)d_in[4];
  const float* ba2  = (const float*)d_in[5];
  const float* Wg1  = (const float*)d_in[6];
  const float* bg1  = (const float*)d_in[7];
  const float* Wg2  = (const float*)d_in[8];
  const float* bg2  = (const float*)d_in[9];
  const float* Wih  = (const float*)d_in[10];
  const float* Whh  = (const float*)d_in[11];
  const float* b_ih = (const float*)d_in[12];
  const float* b_hh = (const float*)d_in[13];
  const int*   aidx = (const int*)d_in[14];
  const int*   step = (const int*)d_in[15];

  const int n = in_sizes[14];            // 131072 alive rows
  const long hidElems = in_sizes[1];     // 262144*128
  const int fs = n / 8;                  // faction size (n_f = 8)
  const int dc = (fs / 4) > 1 ? (fs / 4) : 1;

  float* out = (float*)d_out;
  float* doutH = out + 65;               // new_hiddens region

  uint8_t* wsb = (uint8_t*)d_ws;
  const size_t OFF_WA1 = 0;
  const size_t OFF_WG1 = OFF_WA1 + 48 * 1024;    // 48 frags * 1KB
  const size_t OFF_WA2 = OFF_WG1 + 48 * 1024;
  const size_t OFF_WG2 = OFF_WA2 + 16 * 1024;
  const size_t OFF_WIH = OFF_WG2 + 16 * 1024;
  const size_t OFF_WHH = OFF_WIH + 72 * 1024;
  const size_t OFF_OUT = OFF_WHH + 96 * 1024;    // = 296KB
  const size_t OFF_T   = OFF_OUT + (size_t)n * OUTD * 4;
  const size_t OFF_FS  = OFF_T + (size_t)n * 4;
  const size_t OFF_FM  = OFF_FS + 4096;
  const size_t OFF_GO  = OFF_FM + 4096;
  const size_t OFF_NUM = OFF_GO + 512;
  const size_t OFF_DEN = OFF_NUM + 256;
  const size_t OFF_TS  = OFF_DEN + 4;
  const size_t OFF_TM  = OFF_TS + 4;

  bf16_t* pWa1 = (bf16_t*)(wsb + OFF_WA1);
  bf16_t* pWg1 = (bf16_t*)(wsb + OFF_WG1);
  bf16_t* pWa2 = (bf16_t*)(wsb + OFF_WA2);
  bf16_t* pWg2 = (bf16_t*)(wsb + OFF_WG2);
  bf16_t* pWih = (bf16_t*)(wsb + OFF_WIH);
  bf16_t* pWhh = (bf16_t*)(wsb + OFF_WHH);
  float* out_ws = (float*)(wsb + OFF_OUT);
  float* t_ws   = (float*)(wsb + OFF_T);
  float* fsum   = (float*)(wsb + OFF_FS);
  float* fm     = (float*)(wsb + OFF_FM);
  float* go     = (float*)(wsb + OFF_GO);
  float* numer  = (float*)(wsb + OFF_NUM);
  float* denom  = (float*)(wsb + OFF_DEN);
  float* tsum   = (float*)(wsb + OFF_TS);
  int*   tmaxI  = (int*)(wsb + OFF_TM);

  // full new_hiddens = copy of hiddens, alive rows overwritten by main_k/sync_k
  hipMemcpyAsync(doutH, hid, (size_t)hidElems * sizeof(float),
                 hipMemcpyDeviceToDevice, stream);

  zero_k<<<1, 1024, 0, stream>>>(fsum, numer, denom, tsum, tmaxI);

  pack_w_k<<<96, 256, 0, stream>>>(Wa1, pWa1, 192, 128, 6, 8);
  pack_w_k<<<96, 256, 0, stream>>>(Wg1, pWg1, 192, 128, 6, 8);
  pack_w_k<<<32, 256, 0, stream>>>(Wa2, pWa2, 128, 64, 4, 4);
  pack_w_k<<<32, 256, 0, stream>>>(Wg2, pWg2, 128, 64, 4, 4);
  pack_w_k<<<144, 256, 0, stream>>>(Wih, pWih, 65, 384, 3, 24);
  pack_w_k<<<192, 256, 0, stream>>>(Whh, pWhh, 128, 384, 4, 24);

  main_k<<<n / 128, 256, 0, stream>>>(
      x, hid, ba1, ba2, bg1, bg2, b_ih, b_hh, aidx, pWa1, pWg1, pWa2, pWg2,
      pWih, pWhh, out_ws, t_ws, fsum, tsum, tmaxI, doutH, fs);

  faction_k<<<1, 256, 0, stream>>>(fsum, fm, go, fs);

  sync_k<<<(int)(((long)n * 32 + 255) / 256), 256, 0, stream>>>(
      doutH, aidx, fm, go, step, n, fs, dc);

  combine_k<<<512, 256, 0, stream>>>(out_ws, t_ws, tmaxI, numer, denom, n);

  final_k<<<1, 128, 0, stream>>>(out, numer, denom, tsum, n);
}
